// Seq2SeqModel_54211077210535
// MI455X (gfx1250) — matrix operand order repaired and checked by reading, not compiled, not run on hardware
//
#include <hip/hip_runtime.h>
#include <cstdint>
#include <cstddef>

// Problem dims
#define B_  32
#define S_  64
#define T_  64
#define E_  256
#define H_  512
#define V_  32000
#define G4H 2048   // 4*H
#define KC  768    // E+H
#define MFC 2016   // (T-1)*B rows of the FC GEMM
#define SAP 520    // padded LDS row stride (halves): 1040 B = 65*16 B, bank-offset 4/row

typedef __attribute__((ext_vector_type(16))) __bf16        v16bf;
typedef __attribute__((ext_vector_type(8)))  float         v8f;
typedef __attribute__((ext_vector_type(4)))  unsigned int  u32x4;

union Frag { u32x4 u[2]; v16bf v; };

__device__ __forceinline__ unsigned short f2bf(float f) {
  unsigned u = __float_as_uint(f);
  unsigned r = u + 0x7FFFu + ((u >> 16) & 1u);   // round-to-nearest-even
  return (unsigned short)(r >> 16);
}
__device__ __forceinline__ float sig_(float x) { return 1.0f / (1.0f + __expf(-x)); }

// ---------------- prep kernels ----------------

// Concatenate Wih [2048,256] | Whh [2048,512] -> bf16 [2048,768]
__global__ void make_wcat(const float* __restrict__ Wih, const float* __restrict__ Whh,
                          unsigned short* __restrict__ out) {
  int idx = blockIdx.x * blockDim.x + threadIdx.x;
  if (idx >= G4H * KC) return;
  int n = idx / KC, k = idx % KC;
  float v = (k < E_) ? Wih[n * E_ + k] : Whh[n * H_ + (k - E_)];
  out[idx] = f2bf(v);
}

__global__ void conv_bf16(const float* __restrict__ in, unsigned short* __restrict__ out, int n) {
  int idx = blockIdx.x * blockDim.x + threadIdx.x;
  if (idx < n) out[idx] = f2bf(in[idx]);
}

// x[s][b][e] = bf16(emb[tok[b*64+s]][e])  (tok strided 64 per batch row)
__global__ void gather_emb(const int* __restrict__ tok, const float* __restrict__ emb,
                           unsigned short* __restrict__ out, int steps) {
  int idx = blockIdx.x * blockDim.x + threadIdx.x;
  if (idx >= steps * B_ * E_) return;
  int s = idx / (B_ * E_);
  int rem = idx % (B_ * E_);
  int b = rem / E_, e = rem % E_;
  int t = tok[b * 64 + s];
  out[idx] = f2bf(emb[(size_t)t * E_ + e]);
}

__global__ void zero_u32(unsigned int* __restrict__ p, int n) {
  int idx = blockIdx.x * blockDim.x + threadIdx.x;
  if (idx < n) p[idx] = 0u;
}

// out[b][0][:] = 0
__global__ void zero_t0(float* __restrict__ out) {
  int idx = blockIdx.x * blockDim.x + threadIdx.x;
  if (idx >= B_ * V_) return;
  int b = idx / V_, v = idx % V_;
  out[(size_t)b * T_ * V_ + v] = 0.0f;
}

// ---------------- LSTM step ----------------
// grid = 32 blocks; block b owns H-columns [b*16, b*16+16) across all 4 gates.
// 8 waves: wave = gate*2 + mtile. GEMM via bf16 WMMA, pointwise fused via LDS.
__global__ __launch_bounds__(256) void lstm_step(
    const unsigned short* __restrict__ xemb,   // [32,256] bf16 (this step's embedded input)
    const unsigned short* __restrict__ hin,    // [32,512] bf16
    const unsigned short* __restrict__ Wcat,   // [2048,768] bf16 (Wih|Whh)
    const float* __restrict__ bias,            // [2048]
    float* __restrict__ cbuf,                  // [32,512] f32 (in-place)
    unsigned short* __restrict__ hout,         // [32,512] bf16
    unsigned short* __restrict__ hs_out)       // [32,512] bf16 or nullptr
{
  __shared__ float ldsg[4][32][16];            // 8 KB
  const int tid  = threadIdx.x;
  const int lane = tid & 31;
  const int wave = tid >> 5;
  const int g    = wave >> 1;                  // gate 0..3
  const int mt   = wave & 1;                   // M tile 0..1
  const int nb   = blockIdx.x;                 // 0..31
  const int col0 = g * H_ + nb * 16;           // column in [0,2048)
  const int m0   = mt * 16;

  const int mrow = m0 + (lane & 15);           // A row 0..31
  const int kbA  = (lane >> 4) * 8;            // A K sub-offset (halves)
  const int nB   = col0 + (lane & 15);         // B column = weight row
  const int kbB  = (lane >> 4) * 8;            // B K sub-offset (halves)

  v8f acc = {};
#pragma unroll 4
  for (int k0 = 0; k0 < KC; k0 += 32) {
    const unsigned short* asrc;
    int lda, kk;
    if (k0 < E_) { asrc = xemb; lda = E_; kk = k0; }
    else         { asrc = hin;  lda = H_; kk = k0 - E_; }
    Frag a, b;
    const unsigned short* ap = asrc + (size_t)mrow * lda + kk + kbA;
    a.u[0] = *(const u32x4*)(ap);
    a.u[1] = *(const u32x4*)(ap + 16);
    const unsigned short* bp = Wcat + (size_t)nB * KC + k0 + kbB;
    b.u[0] = *(const u32x4*)(bp);
    b.u[1] = *(const u32x4*)(bp + 16);
    acc = __builtin_amdgcn_wmma_f32_16x16x32_bf16(false, a.v, false, b.v,
                                                  (short)0, acc, false, false);
  }

  // scatter C tile into LDS: VGPR r -> M = r + 8*(lane>=16), N = lane&15
  const int hi8 = (lane >> 4) * 8;
#pragma unroll
  for (int r = 0; r < 8; ++r)
    ldsg[g][m0 + hi8 + r][lane & 15] = acc[r];
  __syncthreads();

  // fused LSTM cell pointwise: 32 rows x 16 cols per block
  const int hbase = nb * 16;
  for (int e = tid; e < 32 * 16; e += 256) {
    int m = e >> 4, nl = e & 15;
    int n = hbase + nl;
    float iv = ldsg[0][m][nl] + bias[n];
    float fv = ldsg[1][m][nl] + bias[H_ + n];
    float gv = ldsg[2][m][nl] + bias[2 * H_ + n];
    float ov = ldsg[3][m][nl] + bias[3 * H_ + n];
    float cp = cbuf[m * H_ + n];
    float cn = sig_(fv) * cp + sig_(iv) * tanhf(gv);
    float hv = sig_(ov) * tanhf(cn);
    cbuf[m * H_ + n] = cn;
    unsigned short hb = f2bf(hv);
    hout[m * H_ + n] = hb;
    if (hs_out) hs_out[m * H_ + n] = hb;
  }
}

// ---------------- FC projection ----------------
// C[2016,32000] = hs[2016,512](bf16) x fc_W^T(bf16) + fc_b, scattered to out[b][t+1][v].
// Block tile 48x128: 8 waves, wave owns 1 N-tile and 3 M-tiles.
// A tile (48x512 bf16, one contiguous 48 KB chunk of hs) is async-staged into LDS
// once per block (global_load_async_to_lds_b128 + s_wait_asynccnt), then all WMMA
// A-fragments come from conflict-free padded LDS; fc_W streams from L2 with prefetch.
__global__ __launch_bounds__(256) void fc_kernel(
    const unsigned short* __restrict__ hs,   // [2016,512] bf16 (t-major: m = t*32+b)
    const unsigned short* __restrict__ Wb,   // [32000,512] bf16
    const float* __restrict__ fcb,           // [32000]
    float* __restrict__ out)                 // [32,64,32000]
{
  __shared__ alignas(16) unsigned short sA[48 * SAP];   // ~48.8 KB
  const int tid   = threadIdx.x;
  const int lane  = tid & 31;
  const int wave  = tid >> 5;
  const int mbase = blockIdx.x * 48;                 // 42 blocks * 48 = 2016
  const int n0    = blockIdx.y * 128 + wave * 16;    // 250 blocks * 128 = 32000

  // ---- async-stage A tile: 48 rows x 1024 B, 16 B per lane per op ----
  {
    const unsigned short* gbase = hs + (size_t)mbase * H_;
#pragma unroll
    for (int it = 0; it < 12; ++it) {
      int chunk = it * 256 + tid;            // 16-byte chunk id, 0..3071
      int row   = chunk >> 6;                // 64 chunks per 1 KB row
      int col   = (chunk & 63) * 8;          // halves within row
      const void* gp = (const void*)(gbase + (size_t)row * H_ + col);
      unsigned lp = (unsigned)(unsigned long long)(const void*)(sA + row * SAP + col);
      asm volatile("global_load_async_to_lds_b128 %0, %1, off"
                   :: "v"(lp), "v"(gp)
                   : "memory");
    }
    asm volatile("s_wait_asynccnt 0x0" ::: "memory");
  }
  __syncthreads();

  const int mlane = lane & 15;
  const int kbA   = (lane >> 4) * 8;
  const int nB    = n0 + (lane & 15);
  const int kbB   = (lane >> 4) * 8;
  const unsigned short* brow = Wb + (size_t)nB * H_;

  v8f acc0 = {}, acc1 = {}, acc2 = {};
#pragma unroll 4
  for (int k0 = 0; k0 < H_; k0 += 32) {
    __builtin_prefetch(brow + k0 + 128, 0, 1);       // global_prefetch_b8 on the fc_W stream
    Frag b;
    b.u[0] = *(const u32x4*)(brow + k0 + kbB);
    b.u[1] = *(const u32x4*)(brow + k0 + kbB + 16);
    const unsigned short* ap = sA + mlane * SAP + k0 + kbA;
    Frag a0, a1, a2;
    a0.u[0] = *(const u32x4*)(ap);
    a0.u[1] = *(const u32x4*)(ap + 16);
    a1.u[0] = *(const u32x4*)(ap + 16 * SAP);
    a1.u[1] = *(const u32x4*)(ap + 16 * SAP + 16);
    a2.u[0] = *(const u32x4*)(ap + 32 * SAP);
    a2.u[1] = *(const u32x4*)(ap + 32 * SAP + 16);
    acc0 = __builtin_amdgcn_wmma_f32_16x16x32_bf16(false, a0.v, false, b.v, (short)0, acc0, false, false);
    acc1 = __builtin_amdgcn_wmma_f32_16x16x32_bf16(false, a1.v, false, b.v, (short)0, acc1, false, false);
    acc2 = __builtin_amdgcn_wmma_f32_16x16x32_bf16(false, a2.v, false, b.v, (short)0, acc2, false, false);
  }

  const int hi8 = (lane >> 4) * 8;
  const int v   = n0 + (lane & 15);
  const float bias = fcb[v];
#pragma unroll
  for (int r = 0; r < 8; ++r) {
    {
      int m = mbase + 0 + hi8 + r;  int ts = m >> 5, bb = m & 31;
      out[((size_t)bb * T_ + ts + 1) * V_ + v] = acc0[r] + bias;
    }
    {
      int m = mbase + 16 + hi8 + r; int ts = m >> 5, bb = m & 31;
      out[((size_t)bb * T_ + ts + 1) * V_ + v] = acc1[r] + bias;
    }
    {
      int m = mbase + 32 + hi8 + r; int ts = m >> 5, bb = m & 31;
      out[((size_t)bb * T_ + ts + 1) * V_ + v] = acc2[r] + bias;
    }
  }
}

// ---------------- launch ----------------
extern "C" void kernel_launch(void* const* d_in, const int* in_sizes, int n_in,
                              void* d_out, int out_size, void* d_ws, size_t ws_size,
                              hipStream_t stream) {
  (void)in_sizes; (void)n_in; (void)out_size; (void)ws_size;
  const int*   src     = (const int*)  d_in[0];
  const int*   tgt     = (const int*)  d_in[1];
  const float* enc_emb = (const float*)d_in[2];
  const float* dec_emb = (const float*)d_in[3];
  const float* enc_Wih = (const float*)d_in[4];
  const float* enc_Whh = (const float*)d_in[5];
  const float* enc_b   = (const float*)d_in[6];
  const float* dec_Wih = (const float*)d_in[7];
  const float* dec_Whh = (const float*)d_in[8];
  const float* dec_b   = (const float*)d_in[9];
  const float* fc_W    = (const float*)d_in[10];
  const float* fc_b    = (const float*)d_in[11];
  float* out = (float*)d_out;

  // workspace layout (256B-aligned slabs)
  char* ws = (char*)d_ws;
  size_t off = 0;
  unsigned short* wcat_enc = (unsigned short*)(ws + off); off += (size_t)G4H * KC * 2;
  unsigned short* wcat_dec = (unsigned short*)(ws + off); off += (size_t)G4H * KC * 2;
  unsigned short* fcWb     = (unsigned short*)(ws + off); off += (size_t)V_ * H_ * 2;
  unsigned short* xsrc     = (unsigned short*)(ws + off); off += (size_t)S_ * B_ * E_ * 2;
  unsigned short* xtgt     = (unsigned short*)(ws + off); off += (size_t)(T_ - 1) * B_ * E_ * 2;
  unsigned short* hsb      = (unsigned short*)(ws + off); off += (size_t)(T_ - 1) * B_ * H_ * 2;
  float*          cbuf     = (float*)(ws + off);          off += (size_t)B_ * H_ * 4;
  unsigned short* hb       = (unsigned short*)(ws + off); off += (size_t)2 * B_ * H_ * 2;

  // prep
  make_wcat<<<(G4H * KC + 255) / 256, 256, 0, stream>>>(enc_Wih, enc_Whh, wcat_enc);
  make_wcat<<<(G4H * KC + 255) / 256, 256, 0, stream>>>(dec_Wih, dec_Whh, wcat_dec);
  conv_bf16<<<(V_ * H_ + 255) / 256, 256, 0, stream>>>(fc_W, fcWb, V_ * H_);
  gather_emb<<<(S_ * B_ * E_ + 255) / 256, 256, 0, stream>>>(src, enc_emb, xsrc, S_);
  gather_emb<<<((T_ - 1) * B_ * E_ + 255) / 256, 256, 0, stream>>>(tgt, dec_emb, xtgt, T_ - 1);
  zero_u32<<<(B_ * H_ + 255) / 256, 256, 0, stream>>>((unsigned int*)cbuf, B_ * H_);
  zero_u32<<<(B_ * H_ + 255) / 256, 256, 0, stream>>>((unsigned int*)hb, B_ * H_); // 2*B*H ushorts
  zero_t0<<<(B_ * V_ + 255) / 256, 256, 0, stream>>>(out);

  const int HBN = B_ * H_;
  int p = 0;
  // encoder: 64 steps
  for (int t = 0; t < S_; ++t, ++p) {
    lstm_step<<<32, 256, 0, stream>>>(xsrc + (size_t)t * B_ * E_,
                                      hb + (size_t)(p & 1) * HBN,
                                      wcat_enc, enc_b, cbuf,
                                      hb + (size_t)((p + 1) & 1) * HBN,
                                      (unsigned short*)nullptr);
  }
  // decoder: 63 steps, h/c carried over from encoder
  for (int t = 0; t < T_ - 1; ++t, ++p) {
    lstm_step<<<32, 256, 0, stream>>>(xtgt + (size_t)t * B_ * E_,
                                      hb + (size_t)(p & 1) * HBN,
                                      wcat_dec, dec_b, cbuf,
                                      hb + (size_t)((p + 1) & 1) * HBN,
                                      hsb + (size_t)t * B_ * H_);
  }
  // FC projection: 42 x 250 blocks of 48x128
  fc_kernel<<<dim3(42, 250), 256, 0, stream>>>(hsb, fcWb, fc_b, out);
}
